// VQVAE_49108656063035
// MI455X (gfx1250) — matrix operand order repaired and checked by reading, not compiled, not run on hardware
//
#include <hip/hip_runtime.h>
#include <hip/hip_bf16.h>

// ---------------------------------------------------------------------------
// VQ-VAE forward, fused, bf16 WMMA (gfx1250 / MI455X)
// B=262144, S=120(pad128), H=256, D=64, K=64 codes
// - Weights pre-swizzled (prep kernel) into WMMA B-fragment order
// - Activations held in LDS in A-fragment order (contiguous 32B frags)
// - Weight slabs DMA'd global->LDS with GLOBAL_LOAD_ASYNC_TO_LDS_B128,
//   double-buffered so the DMA overlaps the WMMA stream (ASYNCcnt tracked)
// ---------------------------------------------------------------------------

typedef __bf16 bf16_t;
typedef __attribute__((ext_vector_type(16))) __bf16 v16bf;
typedef __attribute__((ext_vector_type(8)))  float  v8f;

#define BB 262144
#define NTHREADS 128

// bf16 regions inside d_ws (element offsets). B-layouts are fragment-order:
// idx = ((ks*NT + nt)*32 + lane)*16 + e  <=>  W[ks*32 + (e>>3)*16 + (lane>>4)*8 + (e&7)][nt*16 + (lane&15)]
#define OFF_W1  0LL        // NKS=4 NT=16  [128pad][256]
#define OFF_W2  32768LL    // NKS=8 NT=16  [256][256]
#define OFF_W3  98304LL    // NKS=8 NT=4   [256][64]
#define OFF_F3  114688LL   // NKS=2 NT=16  [64][256]
#define OFF_F4  131072LL   // NKS=8 NT=16  [256][256]
#define OFF_F5  196608LL   // NKS=8 NT=8   [256][128pad]
#define OFF_CB  229376LL   // [64][64] row-major (dist B-frags are row-contiguous)
#define OFF_CBA 233472LL   // [64][4x16] codebook rows in A-fragment chunk order
#define BF_TOTAL 237568LL  // bf16 elements; cnorm (64 f32) follows

__device__ __host__ inline bf16_t tobf(float f) { return (bf16_t)f; }  // native cvt

__device__ inline v8f wmma_bf16(v16bf a, v16bf b, v8f c) {
    return __builtin_amdgcn_wmma_f32_16x16x32_bf16(
        false, a, false, b, (short)0, c, false, false);
}

// ---- async global -> LDS staging (16B per lane-op, ASYNCcnt tracked) -------
__device__ inline void async_stage(const bf16_t* __restrict__ src, unsigned ldsoff,
                                   int tid, int chunks) {
#pragma unroll 1
    for (int i = tid; i < chunks; i += NTHREADS) {
        unsigned long long ga = (unsigned long long)(const void*)(src + i * 8);
        unsigned dst = ldsoff + (unsigned)i * 16u;
        asm volatile("global_load_async_to_lds_b128 %0, %1, off"
                     :: "v"(dst), "v"(ga) : "memory");
    }
}
__device__ inline void wait_async0() {
    asm volatile("s_wait_asynccnt 0x0" ::: "memory");
}

// ---- A fragments from LDS activation buffer (fragment-order layout) --------
// act frag index = (wave*8 + ks)*32 + lane ; each frag = 16 contiguous bf16.
template<int NKS>
__device__ inline void load_afrags(v16bf* afr, const bf16_t* act, int wave, int lane) {
    const v16bf* p = (const v16bf*)act + wave * 256 + lane;
#pragma unroll
    for (int ks = 0; ks < NKS; ++ks) afr[ks] = p[ks * 32];
}

// ---- generic GEMM over K-steps, async double-buffered weight staging -------
template<int NKS, int NDIM>
__device__ inline void gemm_ks(const v16bf* afr, const bf16_t* __restrict__ wg,
                               bf16_t* wbuf, unsigned wbase, v8f* acc,
                               int tid, int lane) {
    constexpr int NT = NDIM / 16;
    constexpr int CH = NT * 64;                 // 16B chunks per K-slab
#pragma unroll
    for (int t = 0; t < NT; ++t) {
        v8f z = {0.f, 0.f, 0.f, 0.f, 0.f, 0.f, 0.f, 0.f};
        acc[t] = z;
    }
    __syncthreads();                            // prior wbuf readers done
    async_stage(wg, wbase, tid, CH);            // prime slab 0
    wait_async0();
    __syncthreads();
    for (int ks = 0; ks < NKS; ++ks) {
        if (ks + 1 < NKS)                       // DMA next slab while we compute
            async_stage(wg + (ks + 1) * NT * 512,
                        wbase + (((ks + 1) & 1) ? 16384u : 0u), tid, CH);
        v16bf a = afr[ks];
        const v16bf* wfr = (const v16bf*)(wbuf + (ks & 1) * 8192);
#pragma unroll
        for (int t = 0; t < NT; ++t)
            acc[t] = wmma_bf16(a, wfr[t * 32 + lane], acc[t]);
        wait_async0();
        __syncthreads();
    }
}

// ---- C-layout -> A-fragment-layout LDS store, bias + relu ------------------
template<int NDIM>
__device__ inline void store_act_relu(const v8f* acc, const float* __restrict__ bias,
                                      bf16_t* act, int wave, int lane) {
    const int n0 = lane & 15, vh = (lane >> 4) * 8;
#pragma unroll
    for (int t = 0; t < NDIM / 16; ++t) {
        int col = t * 16 + n0;
        int ks = col >> 5, kk = col & 31;
        int pos = ((kk >> 4) & 1) * 8 + (kk & 7);
        bf16_t* base = act + (((wave * 8 + ks) * 32) + 16 * ((kk >> 3) & 1)) * 16 + pos;
        float bv = bias[col];
#pragma unroll
        for (int v = 0; v < 8; ++v) {
            float f = fmaxf(acc[t][v] + bv, 0.0f);
            base[(vh + v) * 16] = tobf(f);
        }
    }
}

// ---------------------------------------------------------------------------
// Kernel 0: weights fp32 -> bf16 swizzled into fragment order + codebook prep
// ---------------------------------------------------------------------------
__device__ inline void swz_layer(const float* __restrict__ W, bf16_t* __restrict__ dst,
                                 int NKS, int NT, int Ksrc, int Nsrc,
                                 long long i0, long long st) {
    const long long tot = (long long)NKS * NT * 512;
    for (long long i = i0; i < tot; i += st) {
        int blk = (int)(i >> 9);
        int ks = blk / NT, nt = blk % NT;
        int r = (int)(i & 511);
        int lane = r >> 4, e = r & 15;
        int k = ks * 32 + (e >> 3) * 16 + (lane >> 4) * 8 + (e & 7);
        int n = nt * 16 + (lane & 15);
        float v = (k < Ksrc && n < Nsrc) ? W[k * Nsrc + n] : 0.0f;
        dst[i] = tobf(v);
    }
}

__global__ __launch_bounds__(256) void vqvae_prep(
    const float* __restrict__ w1, const float* __restrict__ w2, const float* __restrict__ w3,
    const float* __restrict__ f3, const float* __restrict__ f4, const float* __restrict__ f5,
    const float* __restrict__ cb, bf16_t* __restrict__ wsb, float* __restrict__ cnorm)
{
    const long long i0 = (long long)blockIdx.x * blockDim.x + threadIdx.x;
    const long long st = (long long)gridDim.x * blockDim.x;
    swz_layer(w1, wsb + OFF_W1, 4, 16, 120, 256, i0, st);
    swz_layer(w2, wsb + OFF_W2, 8, 16, 256, 256, i0, st);
    swz_layer(w3, wsb + OFF_W3, 8, 4, 256, 64, i0, st);
    swz_layer(f3, wsb + OFF_F3, 2, 16, 64, 256, i0, st);
    swz_layer(f4, wsb + OFF_F4, 8, 16, 256, 256, i0, st);
    swz_layer(f5, wsb + OFF_F5, 8, 8, 256, 120, i0, st);
    for (long long i = i0; i < 4096; i += st) wsb[OFF_CB + i] = tobf(cb[i]);
    for (long long i = i0; i < 4096; i += st) {           // A-layout codebook
        int code = (int)(i >> 6), r = (int)(i & 63);
        int ks = (r >> 5) & 1, hih = (r >> 4) & 1, e = r & 15;
        int d = ks * 32 + (e >> 3) * 16 + hih * 8 + (e & 7);
        wsb[OFF_CBA + i] = tobf(cb[code * 64 + d]);
    }
    for (long long i = i0; i < 64; i += st) {
        float s = 0.f;
#pragma unroll 1
        for (int d = 0; d < 64; ++d) { float v = cb[i * 64 + d]; s += v * v; }
        cnorm[i] = s;
    }
}

// ---------------------------------------------------------------------------
// Kernel 1: fused forward. 128 threads = 4 waves, 16 rows/wave, 64 rows/block
// ---------------------------------------------------------------------------
__global__ __launch_bounds__(NTHREADS) void vqvae_main(
    const float* __restrict__ x,
    const float* __restrict__ b1, const float* __restrict__ b2, const float* __restrict__ b3,
    const float* __restrict__ cb,
    const float* __restrict__ b4, const float* __restrict__ b5, const float* __restrict__ b6,
    const bf16_t* __restrict__ w1p, const bf16_t* __restrict__ w2, const bf16_t* __restrict__ w3,
    const bf16_t* __restrict__ f3, const bf16_t* __restrict__ f4, const bf16_t* __restrict__ f5p,
    const bf16_t* __restrict__ cbb, const bf16_t* __restrict__ cbA,
    const float* __restrict__ cnorm,
    float* __restrict__ out_s, float* __restrict__ out_z,
    float* __restrict__ out_emb, float* __restrict__ out_idx)
{
    __shared__ __align__(32) bf16_t act[64 * 256];    // 32 KB, A-fragment order
    __shared__ __align__(32) bf16_t wbuf[2 * 8192];   // 32 KB, 2 K-slabs (ping-pong)
    int* idxbuf = (int*)((char*)wbuf + 24576);        // aliased: dead region at use time

    const int tid  = threadIdx.x;
    const int lane = tid & 31;
    const int wave = tid >> 5;
    const int rowBase = wave * 16;
    const long long gRow0 = (long long)blockIdx.x * 64;
    const unsigned wbase = (unsigned)(unsigned long long)(void*)wbuf;  // LDS byte offset

    v16bf afr[8];
    v8f   acc[16];

    // ---- Encoder L1: [16,128pad] x [128,256], relu ----
    {
        const int hi = (lane >> 4) * 8;
        const float4* xr4 = (const float4*)(x + (gRow0 + rowBase + (lane & 15)) * 120);
#pragma unroll
        for (int ks = 0; ks < 4; ++ks) {
            v16bf a;
            int c0 = ks * 32 + hi;            // chunk 0: k = c0..c0+7 (always in-bounds)
            float4 f0a = xr4[c0 >> 2], f0b = xr4[(c0 >> 2) + 1];
            a[0] = tobf(f0a.x); a[1] = tobf(f0a.y); a[2] = tobf(f0a.z); a[3] = tobf(f0a.w);
            a[4] = tobf(f0b.x); a[5] = tobf(f0b.y); a[6] = tobf(f0b.z); a[7] = tobf(f0b.w);
            int c1 = c0 + 16;                 // chunk 1: fully OOB only for ks=3,hi=8
            if (c1 < 120) {
                float4 f1a = xr4[c1 >> 2], f1b = xr4[(c1 >> 2) + 1];
                a[8]  = tobf(f1a.x); a[9]  = tobf(f1a.y); a[10] = tobf(f1a.z); a[11] = tobf(f1a.w);
                a[12] = tobf(f1b.x); a[13] = tobf(f1b.y); a[14] = tobf(f1b.z); a[15] = tobf(f1b.w);
            } else {
                bf16_t zb = tobf(0.0f);
#pragma unroll
                for (int e = 8; e < 16; ++e) a[e] = zb;
            }
            afr[ks] = a;
        }
        gemm_ks<4, 256>(afr, w1p, wbuf, wbase, acc, tid, lane);
        store_act_relu<256>(acc, b1, act, wave, lane);
    }
    // ---- Encoder L2: [16,256] x [256,256], relu ----
    load_afrags<8>(afr, act, wave, lane);
    gemm_ks<8, 256>(afr, w2, wbuf, wbase, acc, tid, lane);
    store_act_relu<256>(acc, b2, act, wave, lane);
    // ---- Encoder L3: z_e = [16,256] x [256,64] + b3 ----
    load_afrags<8>(afr, act, wave, lane);
    gemm_ks<8, 64>(afr, w3, wbuf, wbase, acc, tid, lane);
    {
        const int n0 = lane & 15, vh = (lane >> 4) * 8;
#pragma unroll
        for (int t = 0; t < 4; ++t) {
            int col = t * 16 + n0;
            int ks = col >> 5, kk = col & 31;
            int pos = ((kk >> 4) & 1) * 8 + (kk & 7);
            bf16_t* base = act + (((wave * 8 + ks) * 32) + 16 * ((kk >> 3) & 1)) * 16 + pos;
            float bv = b3[col];
#pragma unroll
            for (int v = 0; v < 8; ++v) {
                float f = acc[t][v] + bv;
                out_z[(gRow0 + rowBase + vh + v) * 64 + col] = f;  // z_e (fp32)
                base[(vh + v) * 16] = tobf(f);                     // bf16 for quantizer
            }
        }
    }

    // ---- Quantizer: dist = |c|^2 - 2 z.C^T (|z|^2 row-constant, dropped) ----
    __syncthreads();   // wbuf no longer needed as weights
    {
        load_afrags<2>(afr, act, wave, lane);           // z_e tile, K=64
        v8f pacc[4];
#pragma unroll
        for (int t = 0; t < 4; ++t) {
            v8f z = {0.f, 0.f, 0.f, 0.f, 0.f, 0.f, 0.f, 0.f};
            pacc[t] = z;
        }
#pragma unroll
        for (int ks = 0; ks < 2; ++ks) {
#pragma unroll
            for (int t = 0; t < 4; ++t) {
                // B[d][code] = cbb[code][d]; two 8-bf16 chunks per lane (d = 8h.., 16+8h..)
                int code = t * 16 + (lane & 15);
                const bf16_t* p = cbb + code * 64 + ks * 32 + (lane >> 4) * 8;
                v16bf bq;
#pragma unroll
                for (int e = 0; e < 16; ++e) bq[e] = p[(e >> 3) * 16 + (e & 7)];
                pacc[t] = wmma_bf16(afr[ks], bq, pacc[t]);
            }
        }
        float* dist = (float*)wbuf;                     // 64x64 f32 = 16 KB (slab 0)
        const int n0 = lane & 15, vh = (lane >> 4) * 8;
#pragma unroll
        for (int t = 0; t < 4; ++t) {
            int code = t * 16 + n0;
            float cn = cnorm[code];
#pragma unroll
            for (int v = 0; v < 8; ++v)
                dist[(rowBase + vh + v) * 64 + code] = cn - 2.0f * pacc[t][v];
        }
    }
    __syncthreads();
    if (tid < 64) {                                     // per-row argmin + emb copy
        const float* dr = (const float*)wbuf + tid * 64;
        float best = dr[0]; int bi = 0;
#pragma unroll 1
        for (int k = 1; k < 64; ++k) {
            float dv = dr[k];
            if (dv < best) { best = dv; bi = k; }
        }
        idxbuf[tid] = bi;
        long long grow = gRow0 + tid;
        out_idx[grow] = (float)bi;
        const float4* crow = (const float4*)(cb + bi * 64);
        float4*       erow = (float4*)(out_emb + grow * 64);
#pragma unroll
        for (int i = 0; i < 16; ++i) erow[i] = crow[i]; // emb == q (forward value)
    }
    __syncthreads();

    // ---- Decoder L4: z_q (= codebook rows) [16,64] x [64,256], relu ----
    {
        const int code = idxbuf[rowBase + (lane & 15)];
        const v16bf* qa = (const v16bf*)(cbA + code * 64);
        const int hih = lane >> 4;
        afr[0] = qa[0 * 2 + hih];
        afr[1] = qa[1 * 2 + hih];
    }
    gemm_ks<2, 256>(afr, f3, wbuf, wbase, acc, tid, lane);
    store_act_relu<256>(acc, b4, act, wave, lane);
    // ---- Decoder L5: [16,256] x [256,256], relu ----
    load_afrags<8>(afr, act, wave, lane);
    gemm_ks<8, 256>(afr, f4, wbuf, wbase, acc, tid, lane);
    store_act_relu<256>(acc, b5, act, wave, lane);
    // ---- Decoder L6: s_hat = [16,256] x [256,128pad] + b6 (store cols < 120) ----
    load_afrags<8>(afr, act, wave, lane);
    gemm_ks<8, 128>(afr, f5p, wbuf, wbase, acc, tid, lane);
    {
        const int n0 = lane & 15, vh = (lane >> 4) * 8;
#pragma unroll
        for (int t = 0; t < 8; ++t) {
            int col = t * 16 + n0;
            if (col < 120) {
                float bv = b6[col];
#pragma unroll
                for (int v = 0; v < 8; ++v)
                    out_s[(gRow0 + rowBase + vh + v) * 120 + col] = acc[t][v] + bv;
            }
        }
    }
}

// ---------------------------------------------------------------------------
extern "C" void kernel_launch(void* const* d_in, const int* in_sizes, int n_in,
                              void* d_out, int out_size, void* d_ws, size_t ws_size,
                              hipStream_t stream) {
    (void)in_sizes; (void)n_in; (void)out_size; (void)ws_size;
    const float* x   = (const float*)d_in[0];
    const float* w1  = (const float*)d_in[1];
    const float* b1  = (const float*)d_in[2];
    const float* w2  = (const float*)d_in[3];
    const float* b2  = (const float*)d_in[4];
    const float* w3  = (const float*)d_in[5];
    const float* b3  = (const float*)d_in[6];
    const float* cb  = (const float*)d_in[7];
    const float* f3w = (const float*)d_in[8];
    const float* f3b = (const float*)d_in[9];
    const float* f4w = (const float*)d_in[10];
    const float* f4b = (const float*)d_in[11];
    const float* f5w = (const float*)d_in[12];
    const float* f5b = (const float*)d_in[13];

    bf16_t* wsb   = (bf16_t*)d_ws;
    float*  cnorm = (float*)((char*)d_ws + (size_t)(BF_TOTAL * 2));

    float* out_s   = (float*)d_out;
    float* out_z   = out_s   + (long long)BB * 120;
    float* out_emb = out_z   + (long long)BB * 64;
    float* out_idx = out_emb + (long long)BB * 64;

    vqvae_prep<<<256, 256, 0, stream>>>(w1, w2, w3, f3w, f4w, f5w, cb, wsb, cnorm);
    vqvae_main<<<BB / 64, NTHREADS, 0, stream>>>(
        x, b1, b2, b3, cb, f3b, f4b, f5b,
        wsb + OFF_W1, wsb + OFF_W2, wsb + OFF_W3,
        wsb + OFF_F3, wsb + OFF_F4, wsb + OFF_F5,
        wsb + OFF_CB, wsb + OFF_CBA, cnorm,
        out_s, out_z, out_emb, out_idx);
}
